// MultiHeadAttention_28870770163913
// MI455X (gfx1250) — matrix operand order repaired and checked by reading, not compiled, not run on hardware
//
#include <hip/hip_runtime.h>

typedef unsigned short u16;
typedef unsigned int   u32;
typedef unsigned long long u64;
typedef __attribute__((ext_vector_type(16))) __bf16 v16bf;
typedef __attribute__((ext_vector_type(8)))  float  v8f;
typedef __attribute__((ext_vector_type(4)))  unsigned int v4u;
typedef __attribute__((ext_vector_type(8)))  int    v8i;
typedef __attribute__((ext_vector_type(4)))  int    v4i;

#define SEQ     4096
#define DM      768
#define NH      12
#define DK      64

// fragment container: 16 bf16 = 32 bytes = 8 VGPRs
union ABfrag {
    v16bf v;
    u16   u[16];
    uint4 q[2];
};

__device__ __forceinline__ u16 f2bf(float f) {
    u32 u = __float_as_uint(f);
    u32 r = u + 0x7FFFu + ((u >> 16) & 1u);   // round-to-nearest-even
    return (u16)(r >> 16);
}

__device__ __forceinline__ v8f wmma_bf16(v16bf a, v16bf b, v8f c) {
    return __builtin_amdgcn_wmma_f32_16x16x32_bf16(false, a, false, b, (short)0, c, false, false);
}

// async copy 16B global -> LDS, tracked by ASYNCcnt (gfx1250)
__device__ __forceinline__ void async_g2l_b128(u32 lds_off, const void* gaddr) {
    asm volatile("global_load_async_to_lds_b128 %0, %1, off"
                 :: "v"(lds_off), "v"(gaddr) : "memory");
}
#define WAIT_ASYNC(n) asm volatile("s_wait_asynccnt " #n ::: "memory")
#define WAIT_DS0()    asm volatile("s_wait_dscnt 0" ::: "memory")

__device__ __forceinline__ u32 lds_off_u32(const void* p) {
    return (u32)(size_t)p;    // low 32 bits of generic LDS pointer = LDS byte offset
}

// ---- Tensor Data Mover: 2D tile load, data_size = 2 bytes (bf16)
// D# per CDNA5 ISA; groups 2/3 zero (2D tensor).
__device__ __forceinline__ void tdm_load_2d(u32 lds_addr, const void* gaddr,
                                            u32 tensor_d0, u32 tensor_d1,
                                            u32 tile_d0, u32 tile_d1,
                                            u32 stride_d0) {
    const u64 ga = (u64)(size_t)gaddr;
    v4u g0;
    g0.x = 1u;                                            // count=1 (valid user D#)
    g0.y = lds_addr;                                      // LDS byte address
    g0.z = (u32)ga;                                       // global_addr[31:0]
    g0.w = (u32)((ga >> 32) & 0x1FFFFFFu) | (2u << 30);   // addr[56:32] | type=2
    v8i g1;
    g1[0] = (int)(1u << 16);                              // data_size=1 -> 2 bytes
    g1[1] = (int)((tensor_d0 & 0xFFFFu) << 16);           // tensor_dim0[15:0]
    g1[2] = (int)((tensor_d0 >> 16) | ((tensor_d1 & 0xFFFFu) << 16));
    g1[3] = (int)((tensor_d1 >> 16) | (tile_d0 << 16));   // | tile_dim0
    g1[4] = (int)(tile_d1 & 0xFFFFu);                     // tile_dim1 (tile_dim2=0)
    g1[5] = (int)stride_d0;                               // tensor_dim0_stride[31:0]
    g1[6] = 0;
    g1[7] = 0;
    v4i gz; gz[0] = 0; gz[1] = 0; gz[2] = 0; gz[3] = 0;
#if __clang_major__ >= 23
    v8i gz8; for (int i = 0; i < 8; ++i) gz8[i] = 0;
    __builtin_amdgcn_tensor_load_to_lds(g0, g1, gz, gz, gz8, 0);
#else
    __builtin_amdgcn_tensor_load_to_lds(g0, g1, gz, gz, 0);
#endif
}

// ---------------------------------------------------------------- convert
__global__ void cvt_f32_bf16(const float* __restrict__ in, u16* __restrict__ out, int n) {
    int i = blockIdx.x * blockDim.x + threadIdx.x;
    int stride = gridDim.x * blockDim.x;
    for (; i < n; i += stride) out[i] = f2bf(in[i]);
}

// ---------------------------------------------------------------- GEMM
// C[M,N] = A[M,K]*B[K,N] + bias; K=N=768, M=4096.  Per wave 32x64, 8 waves/WG.
// B tile (32x64) staged cooperatively via async->LDS, double buffered.
// MODE 0/2: bf16 [h][l][dk]   MODE 1: bf16 [h][dk][l]   MODE 3: f32 [l][768]
template <int MODE>
__global__ __launch_bounds__(256)
void gemm_wmma(const u16* __restrict__ A, const u16* __restrict__ B,
               const float* __restrict__ bias, void* __restrict__ outp) {
    __shared__ __attribute__((aligned(16))) u16 Bt[2][32 * 64];

    const int tid  = threadIdx.x;
    const int lane = tid & 31;
    const int wave = tid >> 5;
    const int row0 = blockIdx.x * 256 + wave * 32;
    const int col0 = blockIdx.y * 64;

    const int mrow = lane & 15;
    const int kgrp = (lane >> 4) << 3;
    const int krow = (lane & 15) + ((lane >> 4) << 4);
    const int chi  = (lane >> 4) << 3;

    // async B-tile loader: 32 rows x 128B = 256 x 16B chunks, 1 per thread
    const int brow = tid >> 3;
    const int bch  = (tid & 7) << 3;
    auto issueB = [&](int buf, int kk) {
        const u16* gp = B + (size_t)(kk + brow) * DM + col0 + bch;
        async_g2l_b128(lds_off_u32(&Bt[buf][brow * 64 + bch]), gp);
    };

    const v8f zero = {0.f,0.f,0.f,0.f,0.f,0.f,0.f,0.f};
    v8f C[2][4];
    for (int m = 0; m < 2; ++m)
        for (int n = 0; n < 4; ++n) C[m][n] = zero;

    issueB(0, 0);

    ABfrag Af[2];
    for (int m = 0; m < 2; ++m) {
        const u16* ap = A + (size_t)(row0 + m * 16 + mrow) * DM + kgrp;
        Af[m].q[0] = *(const uint4*)(ap);
        Af[m].q[1] = *(const uint4*)(ap + 16);
    }

    for (int kk = 0, it = 0; kk < DM; kk += 32, ++it) {
        const int buf  = it & 1;
        const bool more = (kk + 32) < DM;
        if (more) issueB(buf ^ 1, kk + 32);
        if (more) { WAIT_ASYNC(1); } else { WAIT_ASYNC(0); }
        __syncthreads();                       // Bt[buf] ready for all waves

        ABfrag An[2];
        if (more)
            for (int m = 0; m < 2; ++m) {
                const u16* ap = A + (size_t)(row0 + m * 16 + mrow) * DM + (kk + 32) + kgrp;
                An[m].q[0] = *(const uint4*)(ap);
                An[m].q[1] = *(const uint4*)(ap + 16);
            }

        ABfrag Bf[4];
        for (int n = 0; n < 4; ++n) {
            for (int e = 0; e < 16; ++e) {
                const int k = (e < 8) ? (kgrp + e) : (16 + kgrp + e - 8);
                Bf[n].u[e] = Bt[buf][k * 64 + n * 16 + mrow];
            }
        }
        for (int m = 0; m < 2; ++m)
            for (int n = 0; n < 4; ++n)
                C[m][n] = wmma_bf16(Af[m].v, Bf[n].v, C[m][n]);

        if (more) { Af[0] = An[0]; Af[1] = An[1]; }
        WAIT_DS0();
        __syncthreads();                       // reads done, buffer reusable
    }

    for (int m = 0; m < 2; ++m)
        for (int n = 0; n < 4; ++n) {
            const int c  = col0 + n * 16 + (lane & 15);
            const float bv = bias[c];
            for (int j = 0; j < 8; ++j) {
                const int r = row0 + m * 16 + j + chi;
                const float val = C[m][n][j] + bv;
                if (MODE == 3) {
                    ((float*)outp)[(size_t)r * DM + c] = val;
                } else if (MODE == 1) {
                    ((u16*)outp)[(size_t)c * SEQ + r] = f2bf(val);   // Kt: [c][l]
                } else {
                    ((u16*)outp)[(size_t)(c >> 6) * SEQ * DK + (size_t)r * DK + (c & 63)] = f2bf(val);
                }
            }
        }
}

// ---------------------------------------------------------------- attention
// WG = 4 waves = 64 query rows of one head.  K/V key-block (32) tiles staged
// once per WG via TDM tensor_load_to_lds (wave 0 issues), double buffered.
__global__ __launch_bounds__(128)
void attn_flash(const u16* __restrict__ Qb, const u16* __restrict__ Ktb,
                const u16* __restrict__ Vb, u16* __restrict__ headsb) {
    __shared__ __attribute__((aligned(16))) u16 Ktile[2][64 * 32]; // [dk][key]
    __shared__ __attribute__((aligned(16))) u16 Vtile[2][32 * 64]; // [key][dk]
    __shared__ __attribute__((aligned(16))) u16 plds[4][16 * 32];

    const int tid   = threadIdx.x;
    const int lane  = tid & 31;
    const int wave  = tid >> 5;
    const int h     = blockIdx.y;
    const int qblk  = blockIdx.x * 64;
    const int qbase = qblk + wave * 16;

    const u16* Qh  = Qb  + (size_t)h * SEQ * DK;
    const u16* Kth = Ktb + (size_t)h * DK * SEQ;
    const u16* Vh  = Vb  + (size_t)h * SEQ * DK;

    const int mrow = lane & 15;
    const int kgrp = (lane >> 4) << 3;
    const int krow = (lane & 15) + ((lane >> 4) << 4);
    const int chi  = (lane >> 4) << 3;

    // TDM tile loaders (one descriptor per tile; uniform operands)
    auto issueKV = [&](int buf, int kb) {
        // K tile: tensor Kt[h] is [64][SEQ], tile [64 rows][32 keys]
        tdm_load_2d(lds_off_u32(&Ktile[buf][0]), Kth + kb * 32,
                    SEQ, DK, 32, DK, SEQ);
        // V tile: tensor V[h] is [SEQ][64], tile [32 rows][64 dk]
        tdm_load_2d(lds_off_u32(&Vtile[buf][0]), Vh + (size_t)kb * 32 * DK,
                    DK, SEQ, DK, 32, DK);
    };

    // Q fragments (per wave, two dk slices)
    ABfrag Qf[2];
    for (int s = 0; s < 2; ++s) {
        const u16* qp = Qh + (size_t)(qbase + mrow) * DK + s * 32 + kgrp;
        Qf[s].q[0] = *(const uint4*)(qp);
        Qf[s].q[1] = *(const uint4*)(qp + 16);
    }

    const v8f zero = {0.f,0.f,0.f,0.f,0.f,0.f,0.f,0.f};
    v8f O[4];
    for (int d = 0; d < 4; ++d) O[d] = zero;
    float mrun[8], sden[8];
    for (int j = 0; j < 8; ++j) { mrun[j] = -3.0e38f; sden[j] = 0.f; }

    const float LOG2E = 1.4426950408889634f;
    const int kbmax_wave = (qbase + 15) >> 5;
    const int kbmax_blk  = (qblk + 63) >> 5;

    if (wave == 0) issueKV(0, 0);

    for (int kb = 0; kb <= kbmax_blk; ++kb) {
        const int buf  = kb & 1;
        const bool more = kb < kbmax_blk;
        if (wave == 0) {
            if (more) issueKV(buf ^ 1, kb + 1);
            if (more) __builtin_amdgcn_s_wait_tensorcnt(2);   // current tiles done
            else      __builtin_amdgcn_s_wait_tensorcnt(0);
        }
        __syncthreads();                       // K/V tile[buf] visible to all

        if (kb <= kbmax_wave) {
            // ---- S = Q * K^T  (16 x 32 keys), B frags from LDS
            v8f S[2]; S[0] = zero; S[1] = zero;
            for (int s = 0; s < 2; ++s)
                for (int n = 0; n < 2; ++n) {
                    ABfrag Kf;
                    for (int e = 0; e < 16; ++e) {
                        const int k = (e < 8) ? (kgrp + e) : (16 + kgrp + e - 8);
                        Kf.u[e] = Ktile[buf][(s * 32 + k) * 32 + n * 16 + mrow];
                    }
                    S[n] = wmma_bf16(Qf[s].v, Kf.v, S[n]);
                }
            // ---- scale, causal mask only on diagonal-straddling blocks
            if (kb * 32 + 31 > qbase) {
                for (int n = 0; n < 2; ++n) {
                    const int kidx = kb * 32 + n * 16 + (lane & 15);
                    for (int j = 0; j < 8; ++j) {
                        const int qidx = qbase + j + chi;
                        float v = S[n][j] * 0.125f;
                        if (kidx > qidx) v = -3.0e38f;
                        S[n][j] = v;
                    }
                }
            } else {
                for (int n = 0; n < 2; ++n)
                    for (int j = 0; j < 8; ++j) S[n][j] *= 0.125f;
            }
            // ---- online softmax (row = VGPR j, reduce across 16 lanes)
            for (int j = 0; j < 8; ++j) {
                float mt = fmaxf(S[0][j], S[1][j]);
                for (int off = 1; off < 16; off <<= 1) mt = fmaxf(mt, __shfl_xor(mt, off, 32));
                const float mn = fmaxf(mrun[j], mt);
                const float alpha = exp2f((mrun[j] - mn) * LOG2E);
                const float p0 = exp2f((S[0][j] - mn) * LOG2E);
                const float p1 = exp2f((S[1][j] - mn) * LOG2E);
                float rs = p0 + p1;
                for (int off = 1; off < 16; off <<= 1) rs += __shfl_xor(rs, off, 32);
                sden[j] = sden[j] * alpha + rs;
                mrun[j] = mn;
                for (int d = 0; d < 4; ++d) O[d][j] *= alpha;
                const int prow = j + chi;
                plds[wave][prow * 32 + (lane & 15)]      = f2bf(p0);
                plds[wave][prow * 32 + 16 + (lane & 15)] = f2bf(p1);
            }
            WAIT_DS0();                        // same-wave store->load fence
            ABfrag Pf;
            const u16* pp = &plds[wave][mrow * 32 + kgrp];
            Pf.q[0] = *(const uint4*)(pp);
            Pf.q[1] = *(const uint4*)(pp + 16);
            for (int d = 0; d < 4; ++d) {
                ABfrag Vf;
                for (int e = 0; e < 16; ++e) {
                    const int k = (e < 8) ? (kgrp + e) : (16 + kgrp + e - 8);
                    Vf.u[e] = Vtile[buf][k * 64 + d * 16 + mrow];
                }
                O[d] = wmma_bf16(Pf.v, Vf.v, O[d]);
            }
        }
        WAIT_DS0();
        __syncthreads();                       // all reads done; buffer reusable
    }

    for (int d = 0; d < 4; ++d)
        for (int j = 0; j < 8; ++j) {
            const int row = qbase + j + chi;
            const int col = h * DK + d * 16 + (lane & 15);
            headsb[(size_t)row * DM + col] = f2bf(O[d][j] / sden[j]);
        }
}

// ---------------------------------------------------------------- launch
extern "C" void kernel_launch(void* const* d_in, const int* in_sizes, int n_in,
                              void* d_out, int out_size, void* d_ws, size_t ws_size,
                              hipStream_t stream) {
    const float* x   = (const float*)d_in[0];
    const float* w_q = (const float*)d_in[1];
    const float* b_q = (const float*)d_in[2];
    const float* w_k = (const float*)d_in[3];
    const float* b_k = (const float*)d_in[4];
    const float* w_v = (const float*)d_in[5];
    const float* b_v = (const float*)d_in[6];
    const float* w_o = (const float*)d_in[7];
    const float* b_o = (const float*)d_in[8];

    char* ws = (char*)d_ws;
    size_t off = 0;
    auto alloc = [&](size_t bytes) -> void* {
        void* p = ws + off;
        off = (off + bytes + 255) & ~(size_t)255;
        return p;
    };
    u16* xb  = (u16*)alloc((size_t)SEQ * DM * 2);
    u16* wqb = (u16*)alloc((size_t)DM * DM * 2);
    u16* wkb = (u16*)alloc((size_t)DM * DM * 2);
    u16* wvb = (u16*)alloc((size_t)DM * DM * 2);
    u16* wob = (u16*)alloc((size_t)DM * DM * 2);
    u16* Qb  = (u16*)alloc((size_t)SEQ * DM * 2);
    u16* Ktb = (u16*)alloc((size_t)SEQ * DM * 2);
    u16* Vb  = (u16*)alloc((size_t)SEQ * DM * 2);
    u16* Hb  = (u16*)alloc((size_t)SEQ * DM * 2);

    const int nx = SEQ * DM, nw = DM * DM;
    cvt_f32_bf16<<<(nx + 255) / 256, 256, 0, stream>>>(x,   xb,  nx);
    cvt_f32_bf16<<<(nw + 255) / 256, 256, 0, stream>>>(w_q, wqb, nw);
    cvt_f32_bf16<<<(nw + 255) / 256, 256, 0, stream>>>(w_k, wkb, nw);
    cvt_f32_bf16<<<(nw + 255) / 256, 256, 0, stream>>>(w_v, wvb, nw);
    cvt_f32_bf16<<<(nw + 255) / 256, 256, 0, stream>>>(w_o, wob, nw);

    dim3 ggrid(SEQ / 256, DM / 64);
    gemm_wmma<0><<<ggrid, 256, 0, stream>>>(xb, wqb, b_q, Qb);
    gemm_wmma<1><<<ggrid, 256, 0, stream>>>(xb, wkb, b_k, Ktb);
    gemm_wmma<2><<<ggrid, 256, 0, stream>>>(xb, wvb, b_v, Vb);

    attn_flash<<<dim3(SEQ / 64, NH), 128, 0, stream>>>(Qb, Ktb, Vb, Hb);

    gemm_wmma<3><<<ggrid, 256, 0, stream>>>(Hb, wob, b_o, (float*)d_out);
}
